// SparseSubmConv3d_1563368095915
// MI455X (gfx1250) — matrix operand order repaired and checked by reading, not compiled, not run on hardware
//
#include <hip/hip_runtime.h>
#include <hip/hip_bf16.h>

typedef __attribute__((ext_vector_type(16))) _Float16 v16h;
typedef __attribute__((ext_vector_type(8)))  _Float16 v8h;
typedef __attribute__((ext_vector_type(8)))  float    v8f;

#define KVOL 27
#define CIN  64
#define COUT 64
// 27 k-offsets * 4 cout tiles * 2 cin chunks = 216 B fragments,
// each 32x16 fp16 = 512 halves = 1 KB  ->  221,184 bytes total
#define NFRAG      (KVOL * 8)
#define FRAG_HALF  512
#define B_BYTES    (NFRAG * FRAG_HALF * 2)
#define B_CHUNK16  (B_BYTES / 16)          // 13824 16-byte async chunks

// ---------------------------------------------------------------------------
// Repack weights (KVOL, CIN, COUT) fp16 row-major into WMMA B-fragment
// records in d_ws.  Fragment (k, t, c): B = W[k][c*32 .. c*32+31][t*16 ..
// t*16+15].  This code's B layout (contiguous k, not ISA table order; A matches): lane L holds column n = L%16,
// K group (L/16)*16, element j (0..15) = K group + j.  Stored as
// halves[ frag*512 + lane*16 + j ] so consumers read one aligned 32B v16h.
// ---------------------------------------------------------------------------
__global__ void repack_weights_kernel(const _Float16* __restrict__ w,
                                      _Float16* __restrict__ wsB) {
    int gid = blockIdx.x * blockDim.x + threadIdx.x;
    if (gid >= NFRAG * 32) return;
    int lane = gid & 31;
    int frag = gid >> 5;
    int c    = frag & 1;
    int t    = (frag >> 1) & 3;
    int k    = frag >> 3;
    int n    = t * 16 + (lane & 15);
    int kb   = c * 32 + (lane >> 4) * 16;
    _Float16* dst = wsB + (size_t)frag * FRAG_HALF + (size_t)lane * 16;
    const _Float16* src = w + ((size_t)k * CIN + kb) * COUT + n;
#pragma unroll
    for (int j = 0; j < 16; ++j) dst[j] = src[(size_t)j * COUT];
}

// ---------------------------------------------------------------------------
// Main kernel: 256 threads = 8 waves; whole repacked weight set staged into
// LDS (221 KB of CDNA5's 320 KB/WGP) with GLOBAL_LOAD_ASYNC_TO_LDS_B128
// (ASYNCcnt path, no VGPR round-trip).  Each wave owns 32 output rows (two
// 16-row WMMA tiles), 8 fp32 v8f accumulators, 16 v_wmma_f32_16x16x32_f16
// per kernel offset.
// ---------------------------------------------------------------------------
__global__ __launch_bounds__(256) void
subm_conv_wmma_kernel(const _Float16* __restrict__ features,
                      const int*      __restrict__ indices,
                      const _Float16* __restrict__ wsB,
                      float*          __restrict__ out,
                      int n_pts) {
    extern __shared__ __align__(16) _Float16 Blds[];

    // ---- async stage: global (repacked) -> LDS, 16B per lane per issue ----
    {
        // raw 32-bit LDS byte offset of Blds (addrspace(3) ptrtoint, not the
        // flat aperture address)
        unsigned lds_base = (unsigned)(unsigned long long)
            ((__attribute__((address_space(3))) _Float16*)Blds);
        unsigned long long gbase = (unsigned long long)wsB;
        for (int r = threadIdx.x; r < B_CHUNK16; r += blockDim.x) {
            unsigned           ldst = lds_base + (unsigned)(r * 16);
            unsigned long long gsrc = gbase + (unsigned long long)r * 16u;
            asm volatile("global_load_async_to_lds_b128 %0, %1, off"
                         :: "v"(ldst), "v"(gsrc) : "memory");
        }
        asm volatile("s_wait_asynccnt 0x0" ::: "memory");
    }
    __syncthreads();

    const int lane = threadIdx.x & 31;
    const int m    = lane & 15;       // row within a 16-row tile
    const int h    = lane >> 4;       // half-wave 0/1
    const int klo  = h * 16;          // cin sub-offset for A fragment

    const int wavesPerBlock = blockDim.x >> 5;
    const int wtile = blockIdx.x * wavesPerBlock + (threadIdx.x >> 5);
    const int base  = wtile * 32;     // 32 rows per wave
    if (base >= n_pts) return;        // uniform per wave; barrier already done

    const int  rowgA = base + m;           // tile 0 rows
    const int  rowgB = base + 16 + m;      // tile 1 rows
    const bool okA   = rowgA < n_pts;
    const bool okB   = rowgB < n_pts;

    v8f aA0 = {}, aA1 = {}, aA2 = {}, aA3 = {};   // tile 0 accum (4 cout tiles)
    v8f aB0 = {}, aB1 = {}, aB2 = {}, aB3 = {};   // tile 1 accum
    const v16h vz = {};

    for (int k = 0; k < KVOL; ++k) {
        const size_t kb = (size_t)k * n_pts;
        int  ivA = okA ? indices[kb + rowgA] : -1;
        int  ivB = okB ? indices[kb + rowgB] : -1;
        bool vA  = ivA >= 0;
        bool vB  = ivB >= 0;
        const _Float16* fpA = features + (size_t)(vA ? ivA : 0) * CIN;
        const _Float16* fpB = features + (size_t)(vB ? ivB : 0) * CIN;

        // prefetch next k's index row (streamed once; hide gather latency)
        if (k + 1 < KVOL)
            __builtin_prefetch(indices + kb + n_pts + rowgA, 0, 1);

        // A fragments: per lane, cin {klo..klo+15} contiguous
        // of each 32-wide chunk -> four 16B loads per row tile.
        v16h A0a = __builtin_shufflevector(
            *(const v8h*)(fpA + klo),      *(const v8h*)(fpA + klo + 8),
            0,1,2,3,4,5,6,7,8,9,10,11,12,13,14,15);
        v16h A1a = __builtin_shufflevector(
            *(const v8h*)(fpA + 32 + klo), *(const v8h*)(fpA + 40 + klo),
            0,1,2,3,4,5,6,7,8,9,10,11,12,13,14,15);
        v16h A0b = __builtin_shufflevector(
            *(const v8h*)(fpB + klo),      *(const v8h*)(fpB + klo + 8),
            0,1,2,3,4,5,6,7,8,9,10,11,12,13,14,15);
        v16h A1b = __builtin_shufflevector(
            *(const v8h*)(fpB + 32 + klo), *(const v8h*)(fpB + 40 + klo),
            0,1,2,3,4,5,6,7,8,9,10,11,12,13,14,15);
        A0a = vA ? A0a : vz;  A1a = vA ? A1a : vz;
        A0b = vB ? A0b : vz;  A1b = vB ? A1b : vz;

        // B fragments from LDS: frag = (k*4 + t)*2 + c, 32B per lane.
        const _Float16* bk = Blds + (size_t)k * 8 * FRAG_HALF
                                  + (size_t)lane * 16;
#pragma unroll
        for (int t = 0; t < 4; ++t) {
            v16h B0 = *(const v16h*)(bk + (2 * t + 0) * FRAG_HALF);
            v16h B1 = *(const v16h*)(bk + (2 * t + 1) * FRAG_HALF);
            v8f* dA = (t == 0) ? &aA0 : (t == 1) ? &aA1 : (t == 2) ? &aA2 : &aA3;
            v8f* dB = (t == 0) ? &aB0 : (t == 1) ? &aB1 : (t == 2) ? &aB2 : &aB3;
            *dA = __builtin_amdgcn_wmma_f32_16x16x32_f16(false, A0a, false, B0,
                      (short)0, *dA, false, false);
            *dA = __builtin_amdgcn_wmma_f32_16x16x32_f16(false, A1a, false, B1,
                      (short)0, *dA, false, false);
            *dB = __builtin_amdgcn_wmma_f32_16x16x32_f16(false, A0b, false, B0,
                      (short)0, *dB, false, false);
            *dB = __builtin_amdgcn_wmma_f32_16x16x32_f16(false, A1b, false, B1,
                      (short)0, *dB, false, false);
        }
    }

    // Store.  C/D layout: lane L, VGPR r -> M = (L/16)*8 + r, N = L%16.
    const int rb0 = base + h * 8;        // tile 0
    const int rb1 = base + 16 + h * 8;   // tile 1
#pragma unroll
    for (int r = 0; r < 8; ++r) {
        int row = rb0 + r;
        if (row < n_pts) {
            float* o = out + (size_t)row * COUT + m;
            o[0]  = aA0[r];  o[16] = aA1[r];  o[32] = aA2[r];  o[48] = aA3[r];
        }
        row = rb1 + r;
        if (row < n_pts) {
            float* o = out + (size_t)row * COUT + m;
            o[0]  = aB0[r];  o[16] = aB1[r];  o[32] = aB2[r];  o[48] = aB3[r];
        }
    }
}

extern "C" void kernel_launch(void* const* d_in, const int* in_sizes, int n_in,
                              void* d_out, int out_size, void* d_ws, size_t ws_size,
                              hipStream_t stream) {
    const _Float16* features = reinterpret_cast<const _Float16*>(d_in[0]);
    const int*      indices  = reinterpret_cast<const int*>(d_in[1]);
    const _Float16* weight   = reinterpret_cast<const _Float16*>(d_in[2]);
    float*          out      = reinterpret_cast<float*>(d_out);
    _Float16*       wsB      = reinterpret_cast<_Float16*>(d_ws);

    const int n_pts = in_sizes[0] / CIN;   // N = 100000

    // 1) repack weights into WMMA B-fragment records (221 KB in d_ws)
    {
        int threads = NFRAG * 32;          // 6912
        int blk = 256;
        int grd = (threads + blk - 1) / blk;
        repack_weights_kernel<<<grd, blk, 0, stream>>>(weight, wsB);
    }

    // 2) main WMMA conv: 8 waves/block, 32 rows/wave, B set in 221 KB LDS
    {
        int waves = (n_pts + 31) / 32;     // 3125
        int wavesPerBlock = 8;
        int grd = (waves + wavesPerBlock - 1) / wavesPerBlock;  // 391
        subm_conv_wmma_kernel<<<grd, 256, B_BYTES, stream>>>(features, indices,
                                                             wsB, out, n_pts);
    }
}
